// TokenMergeAttention_11441792877188
// MI455X (gfx1250) — matrix operand order repaired and checked by reading, not compiled, not run on hardware
//
#include <hip/hip_runtime.h>
#include <hip/hip_bf16.h>
#include <math.h>

#define B_   2
#define N_   2048
#define D_   1024
#define NH_  16
#define NKV_ 4
#define HD_  64
#define R_   256
#define NM_  1792      // N_ - R_
#define HALF_ 1024
#define KD_  256       // NKV_*HD_

typedef __bf16 bf16;
typedef __attribute__((ext_vector_type(16))) __bf16 v16bf;
typedef __attribute__((ext_vector_type(8)))  float  v8f;

// ---------------------------------------------------------------------------
// WMMA fragment helpers (wave32; this file's contiguous k order k = hi*16 + j, same in A and B)
// A: 16x32 bf16, row-major source with leading dim lda.
//   lane L (row = L&15, hi = L>>4): elems 0..7  -> K = hi*16 + j
//                                   elems 8..15 -> K = hi*16 + 8 + j
static __device__ __forceinline__ v16bf load_frag_a(const bf16* a, long lda) {
  int lane = threadIdx.x & 31;
  int row  = lane & 15;
  int hi   = lane >> 4;
  const bf16* p = a + (long)row * lda + hi * 16;
  v16bf f;
#pragma unroll
  for (int j = 0; j < 8; ++j) f[j] = p[j];
#pragma unroll
  for (int j = 0; j < 8; ++j) f[8 + j] = p[8 + j];
  return f;
}

// B: 32x16 bf16, supplied TRANSPOSED (bt[n][k], leading dim ldt).
//   lane L (col = L&15, hi = L>>4): elem j -> K = hi*16 + j
static __device__ __forceinline__ v16bf load_frag_bt(const bf16* bt, long ldt) {
  int lane = threadIdx.x & 31;
  int col  = lane & 15;
  int hi   = lane >> 4;
  const bf16* p = bt + (long)col * ldt + hi * 16;
  v16bf f;
#pragma unroll
  for (int j = 0; j < 16; ++j) f[j] = p[j];
  return f;
}

// C/D: 16x16 f32: lane L (col = L&15, hi = L>>4), elem i -> row i + hi*8
static __device__ __forceinline__ void store_frag_c(float* c, long ldc, v8f acc) {
  int lane = threadIdx.x & 31;
  int col  = lane & 15;
  int hi   = lane >> 4;
#pragma unroll
  for (int i = 0; i < 8; ++i) c[(long)(i + hi * 8) * ldc + col] = acc[i];
}

static __device__ __forceinline__ v8f wmma_bf16(v16bf a, v16bf b, v8f c) {
  return __builtin_amdgcn_wmma_f32_16x16x32_bf16(false, a, false, b, (short)0, c,
                                                 false, false);
}

// ---------------------------------------------------------------------------
// Generic batched bf16 GEMM: C[b] = A[b] @ B  (A row-major [M][K], B given
// transposed bf16 [N][K], C f32 [M][N]).
// One 16x64 tile per wave: 4 independent accumulators share one A fragment,
// software-pipelined so next K-slice loads overlap the 4 WMMAs.
__global__ __launch_bounds__(128) void gemm_bf16(
    const bf16* __restrict__ A, const bf16* __restrict__ Bt,
    float* __restrict__ C, int M, int N, int K, int batch,
    long strideA, long strideC) {
  int wave  = blockIdx.x * (blockDim.x >> 5) + (threadIdx.x >> 5);
  int mt = M >> 4, nt = N >> 6;          // 64-wide column groups
  int total = batch * mt * nt;
  if (wave >= total) return;
  int b   = wave / (mt * nt);
  int rem = wave % (mt * nt);
  int mi = rem / nt, ni = rem % nt;
  const bf16* a0 = A + (long)b * strideA + (long)mi * 16 * K;
  const bf16* t0 = Bt + (long)(ni * 64 + 0)  * K;
  const bf16* t1 = Bt + (long)(ni * 64 + 16) * K;
  const bf16* t2 = Bt + (long)(ni * 64 + 32) * K;
  const bf16* t3 = Bt + (long)(ni * 64 + 48) * K;
  v8f acc0 = {0.f, 0.f, 0.f, 0.f, 0.f, 0.f, 0.f, 0.f};
  v8f acc1 = acc0, acc2 = acc0, acc3 = acc0;
  // pipeline prologue
  v16bf af = load_frag_a(a0, K);
  v16bf b0 = load_frag_bt(t0, K);
  v16bf b1 = load_frag_bt(t1, K);
  v16bf b2 = load_frag_bt(t2, K);
  v16bf b3 = load_frag_bt(t3, K);
  for (int k = 32; k < K; k += 32) {
    v16bf afn = load_frag_a(a0 + k, K);
    v16bf b0n = load_frag_bt(t0 + k, K);
    v16bf b1n = load_frag_bt(t1 + k, K);
    v16bf b2n = load_frag_bt(t2 + k, K);
    v16bf b3n = load_frag_bt(t3 + k, K);
    acc0 = wmma_bf16(af, b0, acc0);
    acc1 = wmma_bf16(af, b1, acc1);
    acc2 = wmma_bf16(af, b2, acc2);
    acc3 = wmma_bf16(af, b3, acc3);
    af = afn; b0 = b0n; b1 = b1n; b2 = b2n; b3 = b3n;
  }
  acc0 = wmma_bf16(af, b0, acc0);
  acc1 = wmma_bf16(af, b1, acc1);
  acc2 = wmma_bf16(af, b2, acc2);
  acc3 = wmma_bf16(af, b3, acc3);
  float* c0 = C + (long)b * strideC + (long)mi * 16 * N + (long)ni * 64;
  store_frag_c(c0 + 0,  N, acc0);
  store_frag_c(c0 + 16, N, acc1);
  store_frag_c(c0 + 32, N, acc2);
  store_frag_c(c0 + 48, N, acc3);
}

// ---------------------------------------------------------------------------
// fp32 matching path: metric = x @ Wk.  row uniform per block -> scalar x loads
__global__ void metric_gemm(const float* __restrict__ x, const float* __restrict__ Wk,
                            float* __restrict__ metric) {
  long row = blockIdx.x;                 // 0 .. B_*N_-1
  int c = threadIdx.x;                   // 0 .. 255
  const float* xr = x + row * D_;
  float acc = 0.f;
  for (int k = 0; k < D_; ++k) acc += xr[k] * Wk[(long)k * KD_ + c];
  metric[row * KD_ + c] = acc;
}

// In-place row L2-normalize (rows of length 256); matches a/max(norm,1e-12)
__global__ void normalize_rows(float* __restrict__ metric) {
  long row = blockIdx.x;
  int t = threadIdx.x;
  float v = metric[row * KD_ + t];
  __shared__ float red[256];
  red[t] = v * v;
  __syncthreads();
  for (int o = 128; o > 0; o >>= 1) {
    if (t < o) red[t] += red[t + o];
    __syncthreads();
  }
  float nrm = fmaxf(sqrtf(red[0]), 1e-12f);
  metric[row * KD_ + t] = v / nrm;
}

// best_scores / best_b: for each a-row i, max_j dot(an[i], bn[j]) (first-max tie)
__global__ void best_match(const float* __restrict__ metric,
                           float* __restrict__ best_s, int* __restrict__ best_b) {
  int b = blockIdx.x >> 10;
  int i = blockIdx.x & (HALF_ - 1);
  int t = threadIdx.x;
  const float* base = metric + (long)b * N_ * KD_;
  __shared__ float arow[KD_];
  arow[t] = base[(long)(2 * i) * KD_ + t];
  __syncthreads();
  float bs = -3.0e38f;
  int bj = 0;
  for (int j = t; j < HALF_; j += 256) {
    const float* br = base + (long)(2 * j + 1) * KD_;
    float d = 0.f;
    for (int c = 0; c < KD_; ++c) d += arow[c] * br[c];
    if (d > bs) { bs = d; bj = j; }
  }
  __shared__ float ss[256];
  __shared__ int   jj[256];
  ss[t] = bs; jj[t] = bj;
  __syncthreads();
  for (int o = 128; o > 0; o >>= 1) {
    if (t < o) {
      if (ss[t + o] > ss[t] || (ss[t + o] == ss[t] && jj[t + o] < jj[t])) {
        ss[t] = ss[t + o]; jj[t] = jj[t + o];
      }
    }
    __syncthreads();
  }
  if (t == 0) {
    best_s[(long)b * HALF_ + i] = ss[0];
    best_b[(long)b * HALF_ + i] = jj[0];
  }
}

// Greedy pair selection per batch (== stable argsort(-scores) + scan), plus
// ga/gb, partner map, keep_idx.  One block of 512 threads per batch.
__global__ void select_pairs(const float* __restrict__ best_s,
                             const int* __restrict__ best_b,
                             int* __restrict__ ga, int* __restrict__ gb,
                             int* __restrict__ partner, int* __restrict__ keep_idx) {
  int b = blockIdx.x;
  int tid = threadIdx.x;
  __shared__ float skey[HALF_];
  __shared__ int   sidx[HALF_];
  __shared__ unsigned char used[HALF_];
  __shared__ unsigned char kept[N_];
  __shared__ int sel_a[R_];
  __shared__ int sel_b[R_];
  for (int i = tid; i < HALF_; i += 512) {
    skey[i] = best_s[(long)b * HALF_ + i];
    sidx[i] = i;
    used[i] = 0;
  }
  for (int n = tid; n < N_; n += 512) {
    kept[n] = 1;
    partner[(long)b * N_ + n] = -1;
  }
  __syncthreads();
  // bitonic sort: descending score, ascending index on ties (== stable)
  for (int k = 2; k <= HALF_; k <<= 1) {
    for (int j = k >> 1; j > 0; j >>= 1) {
      for (int i = tid; i < HALF_; i += 512) {
        int ixj = i ^ j;
        if (ixj > i) {
          bool up = ((i & k) == 0);
          float ka = skey[i], kb2 = skey[ixj];
          int ia = sidx[i], ib = sidx[ixj];
          bool first = (ka > kb2) || (ka == kb2 && ia < ib);
          if (up ? !first : first) {
            skey[i] = kb2; skey[ixj] = ka;
            sidx[i] = ib;  sidx[ixj] = ia;
          }
        }
      }
      __syncthreads();
    }
  }
  if (tid == 0) {
    int cnt = 0;
    for (int t = 0; t < HALF_; ++t) {
      int a  = sidx[t];
      int bl = best_b[(long)b * HALF_ + a];
      if (cnt < R_ && !used[bl]) {
        used[bl] = 1;
        sel_a[cnt] = a;
        sel_b[cnt] = bl;
        ++cnt;
      }
    }
    int a0 = (cnt > 0) ? sel_a[0] : 0;
    int b0 = (cnt > 0) ? sel_b[0] : 0;
    for (int r = cnt; r < R_; ++r) { sel_a[r] = a0; sel_b[r] = b0; }
    for (int r = 0; r < R_; ++r) {
      int gA = 2 * sel_a[r];
      int gB = 2 * sel_b[r] + 1;
      ga[(long)b * R_ + r] = gA;
      gb[(long)b * R_ + r] = gB;
      partner[(long)b * N_ + gA] = gB;
      kept[gB] = 0;
    }
    int m = 0;
    for (int n = 0; n < N_ && m < NM_; ++n)
      if (kept[n]) keep_idx[(long)b * NM_ + (m++)] = n;
  }
}

// x_m (merged + gathered) straight to bf16 for the WMMA GEMMs
__global__ void merge_kernel(const float* __restrict__ x, const int* __restrict__ keep_idx,
                             const int* __restrict__ partner, bf16* __restrict__ xmh) {
  int bm = blockIdx.x;            // 0 .. B_*NM_-1
  int b = bm / NM_;
  int src = keep_idx[bm];
  int p = partner[(long)b * N_ + src];
  const float* xr = x + ((long)b * N_ + src) * D_;
  const float* xp = (p >= 0) ? x + ((long)b * N_ + p) * D_ : xr;
  for (int c = threadIdx.x; c < D_; c += 256) {
    float v = xr[c];
    if (p >= 0) v = 0.5f * (v + xp[c]);
    xmh[(long)bm * D_ + c] = (bf16)v;
  }
}

// weight convert + transpose:  W[K][N] f32 -> Wt[N][K] bf16
__global__ void convert_transpose(const float* __restrict__ W, bf16* __restrict__ Wt,
                                  int K, int N) {
  long idx = (long)blockIdx.x * 256 + threadIdx.x;
  if (idx >= (long)K * N) return;
  int k = (int)(idx / N), n = (int)(idx % N);
  Wt[(long)n * K + k] = (bf16)W[idx];
}

__global__ void f32_to_bf16(const float* __restrict__ in, bf16* __restrict__ out, long n) {
  long idx = (long)blockIdx.x * 256 + threadIdx.x;
  if (idx < n) out[idx] = (bf16)in[idx];
}

// RoPE on f32 projections, emit bf16.  t: [B][NM][nheads][64]
__global__ void rope_convert(const float* __restrict__ t, const float* __restrict__ freqs,
                             bf16* __restrict__ out, int nheads) {
  long total = (long)B_ * NM_ * nheads * 32;
  long idx = (long)blockIdx.x * 256 + threadIdx.x;
  if (idx >= total) return;
  int pair = (int)(idx & 31);
  long bmh = idx >> 5;
  int m = (int)((bmh / nheads) % NM_);
  float c = freqs[((long)m * 32 + pair) * 2 + 0];
  float s = freqs[((long)m * 32 + pair) * 2 + 1];
  long base = bmh * 64;
  float t1 = t[base + 2 * pair];
  float t2 = t[base + 2 * pair + 1];
  out[base + 2 * pair]     = (bf16)(t1 * c - t2 * s);
  out[base + 2 * pair + 1] = (bf16)(t1 * s + t2 * c);
}

// v: [B][NM][4][64] f32 -> vt: [B][4][64][NM] bf16 (d-major for B-fragments)
__global__ void v_transpose(const float* __restrict__ v, bf16* __restrict__ vt) {
  long total = (long)B_ * NM_ * KD_;
  long idx = (long)blockIdx.x * 256 + threadIdx.x;
  if (idx >= total) return;
  int d  = (int)(idx & 63);
  int kv = (int)((idx >> 6) & 3);
  int m  = (int)((idx >> 8) % NM_);
  int b  = (int)(idx / ((long)NM_ * KD_));
  vt[(((long)b * NKV_ + kv) * HD_ + d) * NM_ + m] = (bf16)v[idx];
}

// ---------------------------------------------------------------------------
// Flash attention: one wave = 16 query rows of one (b, h).  Streams 32-key
// blocks: S = Q@K^T (4 WMMA), online softmax in LDS, O += P@V (4 WMMA).
__global__ __launch_bounds__(128) void attention_kernel(
    const bf16* __restrict__ qh, const bf16* __restrict__ kh,
    const bf16* __restrict__ vt, float* __restrict__ o) {
  const int nmblk = NM_ / 16;  // 112
  int wave = blockIdx.x * 4 + (threadIdx.x >> 5);
  int b  = wave / (NH_ * nmblk);
  int rem = wave % (NH_ * nmblk);
  int h  = rem / nmblk;
  int mb = rem % nmblk;
  int kv = h >> 2;
  int w = threadIdx.x >> 5;
  int lane = threadIdx.x & 31;
  int col = lane & 15, hi = lane >> 4;

  __shared__ float sS[4][16 * 32];
  __shared__ bf16  sP[4][16 * 32];
  __shared__ float sAl[4][16];
  __shared__ float sM[4][16];
  __shared__ float sL[4][16];

  const bf16* qbase = qh + ((long)b * NM_ + (long)mb * 16) * D_ + h * HD_;
  v16bf qa0 = load_frag_a(qbase, D_);        // d = 0..31
  v16bf qa1 = load_frag_a(qbase + 32, D_);   // d = 32..63
  const bf16* kbase = kh + (long)b * NM_ * KD_ + kv * HD_;
  const bf16* vbase = vt + ((long)b * NKV_ + kv) * HD_ * (long)NM_;

  v8f o0 = {0.f,0.f,0.f,0.f,0.f,0.f,0.f,0.f};
  v8f o1 = o0, o2 = o0, o3 = o0;
  if (lane < 16) { sM[w][lane] = -1e30f; sL[w][lane] = 0.f; }
  __syncthreads();

  for (int kb = 0; kb < NM_; kb += 32) {
    // S tile (16 queries x 32 keys), scaled by 1/sqrt(64)
    v8f s0 = {0.f,0.f,0.f,0.f,0.f,0.f,0.f,0.f};
    v8f s1 = s0;
    s0 = wmma_bf16(qa0, load_frag_bt(kbase + (long)kb * KD_, KD_), s0);
    s0 = wmma_bf16(qa1, load_frag_bt(kbase + (long)kb * KD_ + 32, KD_), s0);
    s1 = wmma_bf16(qa0, load_frag_bt(kbase + (long)(kb + 16) * KD_, KD_), s1);
    s1 = wmma_bf16(qa1, load_frag_bt(kbase + (long)(kb + 16) * KD_ + 32, KD_), s1);
#pragma unroll
    for (int i = 0; i < 8; ++i) {
      sS[w][(i + hi * 8) * 32 + col]      = s0[i] * 0.125f;
      sS[w][(i + hi * 8) * 32 + col + 16] = s1[i] * 0.125f;
    }
    __syncthreads();
    if (lane < 16) {
      int r = lane;
      float mold = sM[w][r];
      float mx = mold;
      for (int j = 0; j < 32; ++j) mx = fmaxf(mx, sS[w][r * 32 + j]);
      float al = expf(mold - mx);
      float sum = 0.f;
      for (int j = 0; j < 32; ++j) {
        float pv = expf(sS[w][r * 32 + j] - mx);
        sP[w][r * 32 + j] = (bf16)pv;
        sum += pv;
      }
      sL[w][r] = sL[w][r] * al + sum;
      sM[w][r] = mx;
      sAl[w][r] = al;
    }
    __syncthreads();
#pragma unroll
    for (int e = 0; e < 8; ++e) {
      float sc = sAl[w][e + hi * 8];
      o0[e] *= sc; o1[e] *= sc; o2[e] *= sc; o3[e] *= sc;
    }
    v16bf pa = load_frag_a(sP[w], 32);
    o0 = wmma_bf16(pa, load_frag_bt(vbase + (long)0  * NM_ + kb, NM_), o0);
    o1 = wmma_bf16(pa, load_frag_bt(vbase + (long)16 * NM_ + kb, NM_), o1);
    o2 = wmma_bf16(pa, load_frag_bt(vbase + (long)32 * NM_ + kb, NM_), o2);
    o3 = wmma_bf16(pa, load_frag_bt(vbase + (long)48 * NM_ + kb, NM_), o3);
    __syncthreads();
  }
  float* obase = o + ((long)b * NM_ + (long)mb * 16) * D_ + h * HD_;
#pragma unroll
  for (int e = 0; e < 8; ++e) {
    int r = e + hi * 8;
    float inv = 1.f / sL[w][r];
    obase[(long)r * D_ + 0  + col] = o0[e] * inv;
    obase[(long)r * D_ + 16 + col] = o1[e] * inv;
    obase[(long)r * D_ + 32 + col] = o2[e] * inv;
    obase[(long)r * D_ + 48 + col] = o3[e] * inv;
  }
}

// unmerge
__global__ void scatter_rows(const float* __restrict__ out_m,
                             const int* __restrict__ keep_idx, float* __restrict__ out) {
  int bm = blockIdx.x;
  int b = bm / NM_;
  int dst = keep_idx[bm];
  for (int c = threadIdx.x; c < D_; c += 256)
    out[((long)b * N_ + dst) * D_ + c] = out_m[(long)bm * D_ + c];
}

__global__ void copy_gb(const int* __restrict__ ga, const int* __restrict__ gb,
                        float* __restrict__ out) {
  int br = blockIdx.x;   // 0 .. B_*R_-1
  int b = br / R_;
  int a = ga[br], g = gb[br];
  for (int c = threadIdx.x; c < D_; c += 256)
    out[((long)b * N_ + g) * D_ + c] = out[((long)b * N_ + a) * D_ + c];
}

// ---------------------------------------------------------------------------
extern "C" void kernel_launch(void* const* d_in, const int* in_sizes, int n_in,
                              void* d_out, int out_size, void* d_ws, size_t ws_size,
                              hipStream_t stream) {
  (void)in_sizes; (void)n_in; (void)ws_size;
  const float* x  = (const float*)d_in[0];
  const float* fr = (const float*)d_in[1];
  const float* Wq = (const float*)d_in[2];
  const float* Wk = (const float*)d_in[3];
  const float* Wv = (const float*)d_in[4];
  const float* Wo = (const float*)d_in[5];
  float* out = (float*)d_out;

  char* p = (char*)d_ws;
  auto alloc = [&](size_t bytes) -> char* {
    char* r = p;
    p += (bytes + 255) & ~(size_t)255;
    return r;
  };
  float* metric  = (float*)alloc((size_t)B_ * N_ * KD_ * 4);
  float* best_s  = (float*)alloc((size_t)B_ * HALF_ * 4);
  int*   best_b  = (int*)  alloc((size_t)B_ * HALF_ * 4);
  int*   ga      = (int*)  alloc((size_t)B_ * R_ * 4);
  int*   gb      = (int*)  alloc((size_t)B_ * R_ * 4);
  int*   keepidx = (int*)  alloc((size_t)B_ * NM_ * 4);
  int*   partner = (int*)  alloc((size_t)B_ * N_ * 4);
  bf16*  Wqt     = (bf16*) alloc((size_t)D_ * D_ * 2);
  bf16*  Wkt     = (bf16*) alloc((size_t)D_ * KD_ * 2);
  bf16*  Wvt     = (bf16*) alloc((size_t)D_ * KD_ * 2);
  bf16*  Wot     = (bf16*) alloc((size_t)D_ * D_ * 2);
  bf16*  xmh     = (bf16*) alloc((size_t)B_ * NM_ * D_ * 2);
  float* qtmp    = (float*)alloc((size_t)B_ * NM_ * D_ * 4);
  float* ktmp    = (float*)alloc((size_t)B_ * NM_ * KD_ * 4);
  float* vtmp    = (float*)alloc((size_t)B_ * NM_ * KD_ * 4);
  bf16*  qhb     = (bf16*) alloc((size_t)B_ * NM_ * D_ * 2);
  bf16*  khb     = (bf16*) alloc((size_t)B_ * NM_ * KD_ * 2);
  bf16*  vtb     = (bf16*) alloc((size_t)B_ * NM_ * KD_ * 2);
  float* obuf    = (float*)alloc((size_t)B_ * NM_ * D_ * 4);
  bf16*  ohb     = (bf16*) alloc((size_t)B_ * NM_ * D_ * 2);
  float* outm    = (float*)alloc((size_t)B_ * NM_ * D_ * 4);

  // 1. weights -> bf16 transposed
  convert_transpose<<<dim3((D_ * D_ + 255) / 256), 256, 0, stream>>>(Wq, Wqt, D_, D_);
  convert_transpose<<<dim3((D_ * KD_ + 255) / 256), 256, 0, stream>>>(Wk, Wkt, D_, KD_);
  convert_transpose<<<dim3((D_ * KD_ + 255) / 256), 256, 0, stream>>>(Wv, Wvt, D_, KD_);
  convert_transpose<<<dim3((D_ * D_ + 255) / 256), 256, 0, stream>>>(Wo, Wot, D_, D_);

  // 2. matching path (fp32 for selection stability)
  metric_gemm<<<dim3(B_ * N_), 256, 0, stream>>>(x, Wk, metric);
  normalize_rows<<<dim3(B_ * N_), 256, 0, stream>>>(metric);
  best_match<<<dim3(B_ * HALF_), 256, 0, stream>>>(metric, best_s, best_b);
  select_pairs<<<dim3(B_), 512, 0, stream>>>(best_s, best_b, ga, gb, partner, keepidx);

  // 3. merge tokens -> bf16
  merge_kernel<<<dim3(B_ * NM_), 256, 0, stream>>>(x, keepidx, partner, xmh);

  // 4. QKV projections (bf16 WMMA, 16x64 tiles per wave)
  {
    long sa = (long)NM_ * D_;
    int tq = B_ * (NM_ / 16) * (D_ / 64);    // 3584 waves
    int tk = B_ * (NM_ / 16) * (KD_ / 64);   // 896 waves
    gemm_bf16<<<dim3((tq + 3) / 4), 128, 0, stream>>>(
        xmh, Wqt, qtmp, NM_, D_, D_, B_, sa, (long)NM_ * D_);
    gemm_bf16<<<dim3((tk + 3) / 4), 128, 0, stream>>>(
        xmh, Wkt, ktmp, NM_, KD_, D_, B_, sa, (long)NM_ * KD_);
    gemm_bf16<<<dim3((tk + 3) / 4), 128, 0, stream>>>(
        xmh, Wvt, vtmp, NM_, KD_, D_, B_, sa, (long)NM_ * KD_);
  }

  // 5. RoPE + bf16, V transposed d-major
  rope_convert<<<dim3((int)(((long)B_ * NM_ * NH_ * 32 + 255) / 256)), 256, 0, stream>>>(
      qtmp, fr, qhb, NH_);
  rope_convert<<<dim3((int)(((long)B_ * NM_ * NKV_ * 32 + 255) / 256)), 256, 0, stream>>>(
      ktmp, fr, khb, NKV_);
  v_transpose<<<dim3((int)(((long)B_ * NM_ * KD_ + 255) / 256)), 256, 0, stream>>>(vtmp, vtb);

  // 6. attention (exact grid: 896 blocks * 4 waves = B*NH*(NM/16) wave tiles)
  attention_kernel<<<dim3(B_ * NH_ * (NM_ / 16) / 4), 128, 0, stream>>>(qhb, khb, vtb, obuf);

  // 7. output projection
  f32_to_bf16<<<dim3((int)(((long)B_ * NM_ * D_ + 255) / 256)), 256, 0, stream>>>(
      obuf, ohb, (long)B_ * NM_ * D_);
  {
    int to = B_ * (NM_ / 16) * (D_ / 64);
    gemm_bf16<<<dim3((to + 3) / 4), 128, 0, stream>>>(
        ohb, Wot, outm, NM_, D_, D_, B_, (long)NM_ * D_, (long)NM_ * D_);
  }

  // 8. unmerge
  hipMemsetAsync(out, 0, (size_t)out_size * sizeof(float), stream);
  scatter_rows<<<dim3(B_ * NM_), 256, 0, stream>>>(outm, keepidx, out);
  copy_gb<<<dim3(B_ * R_), 256, 0, stream>>>(ga, gb, out);
}
